// GAT_Quant_13486197310314
// MI455X (gfx1250) — matrix operand order repaired and checked by reading, not compiled, not run on hardware
//
#include <hip/hip_runtime.h>
#include <hip/hip_bf16.h>
#include <math.h>

#define IN_F 512
#define NEG_SLOPE 0.2f

typedef __attribute__((ext_vector_type(16))) __bf16 v16bf;
typedef __attribute__((ext_vector_type(8)))  float  v8f;

union BF16x16 { v16bf v; uint4 q[2]; };

// pack two fp32 into one dword of two RNE-rounded bf16 (lo, hi)
__device__ __forceinline__ unsigned int pkbf(float lo, float hi_) {
  unsigned int xl = __float_as_uint(lo), xh = __float_as_uint(hi_);
  xl += 0x7FFFu + ((xl >> 16) & 1u);
  xh += 0x7FFFu + ((xh >> 16) & 1u);
  return (xl >> 16) | (xh & 0xFFFF0000u);
}

__device__ __forceinline__ uint4  ldq(const unsigned short* p) { return *(const uint4*)p; }
__device__ __forceinline__ float4 ld4(const float* p)          { return *(const float4*)p; }

__device__ __forceinline__ void atomicMaxFloat(float* addr, float val) {
  // monotonic int/uint punning; valid with -inf init and mixed signs
  if (val >= 0.0f) atomicMax((int*)addr, __float_as_int(val));
  else             atomicMin((unsigned int*)addr, __float_as_uint(val));
}

__device__ __forceinline__ void edge_nodes(const int* __restrict__ src,
                                           const int* __restrict__ dst,
                                           int e, int Eo, int& s, int& d) {
  if (e < Eo) { s = src[e]; d = dst[e]; }
  else        { s = e - Eo; d = s; }      // implicit self-loops
}

__global__ void fill_f32(float* __restrict__ p, float v, int n) {
  int t = blockIdx.x * blockDim.x + threadIdx.x;
  if (t < n) p[t] = v;
}

// streaming fp32 -> packed bf16 (RNE); n4 = count/4
__global__ void cvt_bf16(const float* __restrict__ src, unsigned int* __restrict__ dst, int n4) {
  int t = blockIdx.x * blockDim.x + threadIdx.x;
  if (t >= n4) return;
  float4 v = ((const float4*)src)[t];
  uint2 o; o.x = pkbf(v.x, v.y); o.y = pkbf(v.z, v.w);
  ((uint2*)dst)[t] = o;
}

// C[M,O] = A[M,K] @ B[O,K]^T, A/B pre-packed bf16, v_wmma_f32_16x16x32_bf16.
// Wave computes a 16 x (NT*16) strip; one A fragment feeds NT WMMAs per K-step.
// Inner loop is pure b128 loads + WMMA (no conversion ALU).
// grid = (ceil(O/(64*NT)), ceil(M/16)), block = 128 (4 waves).
template <int K, int NT>
__global__ __launch_bounds__(128)
void gemm_wmma_bf16(const unsigned short* __restrict__ A,
                    const unsigned short* __restrict__ Bw,
                    float* __restrict__ C, int M, int O) {
  const int lane = threadIdx.x & 31;
  const int wave = threadIdx.x >> 5;
  const int l16  = lane & 15;
  const bool hi  = lane >= 16;
  const int m0 = blockIdx.y << 4;
  const int nw = (blockIdx.x * 4 + wave) * (NT * 16);
  if (m0 >= M || nw >= O) return;                  // wave-uniform: EXEC stays full
  const int ra = (m0 + l16 < M) ? (m0 + l16) : (M - 1);
  const unsigned short* arow = A + (size_t)ra * K;
  const unsigned short* brow[NT];
#pragma unroll
  for (int t = 0; t < NT; ++t)
    brow[t] = Bw + (size_t)(nw + t * 16 + l16) * K;
  const int ka = hi ? 16 : 0;   // A: lanes 0-15 K 0..15, hi lanes K 16..31
  const int kb = hi ? 16 : 0;   // B: lanes 0-15 K 0..15, hi lanes K 16..31

  v8f acc[NT] = {};
#pragma unroll 4
  for (int k0 = 0; k0 < K; k0 += 32) {
    BF16x16 a;
    a.q[0] = ldq(arow + k0 + ka);        // VGPR0-3: K ka..ka+7
    a.q[1] = ldq(arow + k0 + ka + 8);    // VGPR4-7: K ka+8..ka+15
#pragma unroll
    for (int t = 0; t < NT; ++t) {
      BF16x16 b;
      b.q[0] = ldq(brow[t] + k0 + kb);       // K kb..kb+7
      b.q[1] = ldq(brow[t] + k0 + kb + 8);   // K kb+8..kb+15
      acc[t] = __builtin_amdgcn_wmma_f32_16x16x32_bf16(false, a.v, false, b.v,
                                                       (short)0, acc[t], false, false);
    }
  }

  // C layout: VGPR v -> row m0+v(+8 for hi lanes), col = n-tile + l16
  const int mrow = m0 + (hi ? 8 : 0);
  if (m0 + 16 <= M) {                              // uniform fast path, no guards
#pragma unroll
    for (int t = 0; t < NT; ++t) {
      float* cp = C + (size_t)mrow * O + nw + t * 16 + l16;
#pragma unroll
      for (int v = 0; v < 8; ++v) cp[(size_t)v * O] = acc[t][v];
    }
  } else {
#pragma unroll
    for (int t = 0; t < NT; ++t)
#pragma unroll
      for (int v = 0; v < 8; ++v) {
        int m = mrow + v;
        if (m < M) C[(size_t)m * O + nw + t * 16 + l16] = acc[t][v];
      }
  }
}

// per-(node,head) attention logit halves: as = <h, att_src>, ad = <h, att_dst>
template <int HH, int CC>
__global__ void attn_dots(const float* __restrict__ h, const float* __restrict__ att_s,
                          const float* __restrict__ att_d, float* __restrict__ as_,
                          float* __restrict__ ad_, int n) {
  int t = blockIdx.x * blockDim.x + threadIdx.x;
  if (t >= n * HH) return;
  int hh = t & (HH - 1);
  const float* row = h + (size_t)t * CC;           // t == node*HH + hh; rows contiguous
  float ss = 0.f, dd = 0.f;
#pragma unroll 4
  for (int c = 0; c < CC; c += 4) {
    float4 v = ld4(row + c);
    const float* s4 = att_s + hh * CC + c;
    const float* d4 = att_d + hh * CC + c;
    ss += v.x * s4[0] + v.y * s4[1] + v.z * s4[2] + v.w * s4[3];
    dd += v.x * d4[0] + v.y * d4[1] + v.z * d4[2] + v.w * d4[3];
  }
  as_[t] = ss; ad_[t] = dd;
}

template <int HH>
__global__ void edge_max(const int* __restrict__ src, const int* __restrict__ dst,
                         int Eo, int n, const float* __restrict__ as_,
                         const float* __restrict__ ad_, float* __restrict__ mbuf) {
  int t = blockIdx.x * blockDim.x + threadIdx.x;
  if (t >= (Eo + n) * HH) return;
  int e = t / HH, hh = t & (HH - 1);
  int s, d; edge_nodes(src, dst, e, Eo, s, d);
  float v = as_[s * HH + hh] + ad_[d * HH + hh];
  v = (v >= 0.f) ? v : NEG_SLOPE * v;
  atomicMaxFloat(&mbuf[d * HH + hh], v);
}

template <int HH>
__global__ void edge_sum(const int* __restrict__ src, const int* __restrict__ dst,
                         int Eo, int n, const float* __restrict__ as_,
                         const float* __restrict__ ad_, const float* __restrict__ mbuf,
                         float* __restrict__ sbuf) {
  int t = blockIdx.x * blockDim.x + threadIdx.x;
  if (t >= (Eo + n) * HH) return;
  int e = t / HH, hh = t & (HH - 1);
  int s, d; edge_nodes(src, dst, e, Eo, s, d);
  float v = as_[s * HH + hh] + ad_[d * HH + hh];
  v = (v >= 0.f) ? v : NEG_SLOPE * v;
  atomicAdd(&sbuf[d * HH + hh], __expf(v - mbuf[d * HH + hh]));
}

// one wave per edge; lane covers (HH*CC)/32 contiguous channels (stays inside one head)
template <int HH, int CC>
__global__ void edge_agg(const int* __restrict__ src, const int* __restrict__ dst,
                         int Eo, int n, const float* __restrict__ as_,
                         const float* __restrict__ ad_, const float* __restrict__ mbuf,
                         const float* __restrict__ sbuf, const float* __restrict__ hfeat,
                         float* __restrict__ outb) {
  constexpr int F   = HH * CC;
  constexpr int CPL = F / 32;        // 8 for layer1, 2 for layer2
  int lane = threadIdx.x & 31;
  int e = blockIdx.x * (blockDim.x >> 5) + (threadIdx.x >> 5);
  if (e >= Eo + n) return;
  int s, d; edge_nodes(src, dst, e, Eo, s, d);
  const int f0 = lane * CPL;
  const int hh = f0 / CC;
  float v = as_[s * HH + hh] + ad_[d * HH + hh];
  v = (v >= 0.f) ? v : NEG_SLOPE * v;
  const float alpha = __expf(v - mbuf[d * HH + hh]) / (sbuf[d * HH + hh] + 1e-16f);
  const float* hrow = hfeat + (size_t)s * F + f0;
  float*       orow = outb  + (size_t)d * F + f0;
#pragma unroll
  for (int j = 0; j < CPL; ++j)
    atomicAdd(&orow[j], hrow[j] * alpha);
}

template <int F, int DO_ELU>
__global__ void bias_act(float* __restrict__ x, const float* __restrict__ b, int n) {
  int t = blockIdx.x * blockDim.x + threadIdx.x;
  if (t >= n * F) return;
  float v = x[t] + b[t & (F - 1)];
  if (DO_ELU) v = (v > 0.f) ? v : (__expf(v) - 1.f);
  x[t] = v;
}

extern "C" void kernel_launch(void* const* d_in, const int* in_sizes, int n_in,
                              void* d_out, int out_size, void* d_ws, size_t ws_size,
                              hipStream_t stream) {
  (void)n_in; (void)out_size; (void)ws_size;
  const float* x      = (const float*)d_in[0];
  const int*   ei     = (const int*)d_in[1];
  const float* W1     = (const float*)d_in[2];
  const float* att_s1 = (const float*)d_in[3];
  const float* att_d1 = (const float*)d_in[4];
  const float* b1     = (const float*)d_in[5];
  const float* W2     = (const float*)d_in[6];
  const float* att_s2 = (const float*)d_in[7];
  const float* att_d2 = (const float*)d_in[8];
  const float* b2     = (const float*)d_in[9];
  float* out = (float*)d_out;

  const int N    = in_sizes[0] / IN_F;   // 100000
  const int E    = in_sizes[1] / 2;      // 1600000
  const int Etot = E + N;                // + self loops
  const int* src = ei;
  const int* dst = ei + E;

  float* w = (float*)d_ws;
  float* h1   = w; w += (size_t)N * 256;
  float* out1 = w; w += (size_t)N * 256;   // x2 after bias+ELU; x_bf lives here first
  float* as1  = w; w += (size_t)N * 8;
  float* ad1  = w; w += (size_t)N * 8;
  float* m1   = w; w += (size_t)N * 8;
  float* s1   = w; w += (size_t)N * 8;
  float* as2  = w; w += N;
  float* ad2  = w; w += N;
  float* m2   = w; w += N;
  float* s2   = w; w += N;
  unsigned short* W1bf = (unsigned short*)w;       // 256*512 bf16
  unsigned short* W2bf = W1bf + 256 * 512;         // 64*256 bf16

  // x_bf: N*512 bf16 == N*1024 B == out1 region (dead until zeroed after GEMM1)
  unsigned short* xbf  = (unsigned short*)out1;
  // layer-2: h2 (N*64 f32) in low part of dead h1 region; x2_bf right after (disjoint)
  float*          h2   = h1;
  unsigned short* x2bf = (unsigned short*)(h1 + (size_t)N * 64);

  auto cdiv = [](int a, int b) { return (a + b - 1) / b; };

  // ---- init small accumulators (d_out/d_ws are poisoned by harness) ----
  fill_f32<<<cdiv(N * 8, 256), 256, 0, stream>>>(m1, -INFINITY, N * 8);
  fill_f32<<<cdiv(N * 8, 256), 256, 0, stream>>>(s1, 0.f, N * 8);
  fill_f32<<<cdiv(N, 256), 256, 0, stream>>>(m2, -INFINITY, N);
  fill_f32<<<cdiv(N, 256), 256, 0, stream>>>(s2, 0.f, N);
  fill_f32<<<cdiv(N * 64, 256), 256, 0, stream>>>(out, 0.f, N * 64);

  // ---- layer 1: GAT(512 -> 8 heads x 32, concat) ----
  cvt_bf16<<<cdiv(N * IN_F / 4, 256), 256, 0, stream>>>(x, (unsigned int*)xbf, N * IN_F / 4);
  cvt_bf16<<<cdiv(256 * IN_F / 4, 256), 256, 0, stream>>>(W1, (unsigned int*)W1bf, 256 * IN_F / 4);
  gemm_wmma_bf16<IN_F, 4><<<dim3(1, cdiv(N, 16)), 128, 0, stream>>>(xbf, W1bf, h1, N, 256);
  fill_f32<<<cdiv(N * 256, 256), 256, 0, stream>>>(out1, 0.f, N * 256);  // kills x_bf (dead)
  attn_dots<8, 32><<<cdiv(N * 8, 256), 256, 0, stream>>>(h1, att_s1, att_d1, as1, ad1, N);
  edge_max<8><<<cdiv(Etot * 8, 256), 256, 0, stream>>>(src, dst, E, N, as1, ad1, m1);
  edge_sum<8><<<cdiv(Etot * 8, 256), 256, 0, stream>>>(src, dst, E, N, as1, ad1, m1, s1);
  edge_agg<8, 32><<<cdiv(Etot, 8), 256, 0, stream>>>(src, dst, E, N, as1, ad1, m1, s1, h1, out1);
  bias_act<256, 1><<<cdiv(N * 256, 256), 256, 0, stream>>>(out1, b1, N);  // + b1, ELU

  // ---- layer 2: GAT(256 -> 1 head x 64, mean==identity) ----
  cvt_bf16<<<cdiv(N * 256 / 4, 256), 256, 0, stream>>>(out1, (unsigned int*)x2bf, N * 256 / 4);
  cvt_bf16<<<cdiv(64 * 256 / 4, 256), 256, 0, stream>>>(W2, (unsigned int*)W2bf, 64 * 256 / 4);
  gemm_wmma_bf16<256, 1><<<dim3(1, cdiv(N, 16)), 128, 0, stream>>>(x2bf, W2bf, h2, N, 64);
  attn_dots<1, 64><<<cdiv(N, 256), 256, 0, stream>>>(h2, att_s2, att_d2, as2, ad2, N);
  edge_max<1><<<cdiv(Etot, 256), 256, 0, stream>>>(src, dst, E, N, as2, ad2, m2);
  edge_sum<1><<<cdiv(Etot, 256), 256, 0, stream>>>(src, dst, E, N, as2, ad2, m2, s2);
  edge_agg<1, 64><<<cdiv(Etot, 8), 256, 0, stream>>>(src, dst, E, N, as2, ad2, m2, s2, h2, out);
  bias_act<64, 0><<<cdiv(N * 64, 256), 256, 0, stream>>>(out, b2, N);     // + b2
}
